// PANLoss_14791867367735
// MI455X (gfx1250) — compile-verified, matrix operand order checked by reading, not run on hardware
//
#include <hip/hip_runtime.h>
#include <math.h>

// ---------------------------------------------------------------------------
// PAN loss for MI455X (gfx1250, wave32).
// Memory-bound: ~220MB over 2 streaming passes ~= 10us at 23.3 TB/s (inputs
// fit in 192MB L2, so pass 2 largely hits L2).
// Centroid/count segmented sums are mapped to V_WMMA_F32_16X16X32_F16:
//   A[16x32] f16 : rows 0-7 = kernel one-hot, rows 8-15 = text one-hot (exact)
//   B[32x16] f16 : cols 0-3 = sim channels, col 4 = ones (counts)
//   C[16x16] f32 : masked sim sums + kcount + tcount, accumulated in f32.
// Fragment build: 3 ds_bpermute + ~6 VALU per element (one-hot packed in a
// single int; sim packed as two f16 pairs via v_cvt_pk_f16_f32).
// ---------------------------------------------------------------------------

typedef __attribute__((ext_vector_type(16))) _Float16 v16h;
typedef __attribute__((ext_vector_type(2)))  __fp16   v2fp16;   // cvt_pkrtz return type
typedef __attribute__((ext_vector_type(8)))  float    v8f;

#define NPIX   409600
#define NBATCH 8
#define HBINS  1024
#define HSCALE 64.0f          // bins per unit score, range [-8, 8)

// per-batch workspace layout (32-bit words)
#define OFF_KSUM 0            // 8*4 floats: masked sim sums per instance
#define OFF_KCNT 32           // 8 floats
#define OFF_TCNT 40           // 8 floats
#define OFF_POS  48           // int
#define OFF_POSM 49           // int
#define OFF_NEG  50           // int
#define OFF_HIST 64           // 1024 ints
#define OFF_G    1088         // 8*4 floats: centroids
#define OFF_THR  1120         // float
#define OFF_FLB  1121         // int
#define OFF_NV   1122         // int
#define OFF_LDIS 1123         // float
#define OFF_AGG  1124         // 8 floats
#define OFF_DT   1132         // 3 floats (a,b,c) dice text
#define OFF_DK   1136         // 3 floats (a,b,c) dice kernel
#define WSTRIDE  1280         // words per batch

union Frag16 { v16h v; unsigned short u[16]; };
union H2U    { v2fp16 h; unsigned int u; };

__global__ void pan_zero(float* ws) {
    int i = blockIdx.x * blockDim.x + threadIdx.x;  // exactly NBATCH*WSTRIDE threads
    ws[i] = 0.0f;
}

// Pass 1: WMMA segmented sums/counts + OHEM stats + score histogram.
__global__ void pan_pass1(const float* __restrict__ outs, const int* __restrict__ labs,
                          const float* __restrict__ tmg, float* __restrict__ ws) {
    const int b    = blockIdx.y;
    const int tid  = threadIdx.x;
    const int wave = tid >> 5, lane = tid & 31;
    const int Mrow = lane & 15, half = lane >> 4;   // A-row / B-col = lane%16

    const float* txt  = outs + (size_t)(b * 6 + 0) * NPIX;
    const float* simB = outs + (size_t)(b * 6 + 2) * NPIX;
    const int*   gtT  = labs + (size_t)(b * 2 + 0) * NPIX;
    const int*   gtK  = labs + (size_t)(b * 2 + 1) * NPIX;
    const float* tmb  = tmg + (size_t)b * NPIX;
    float*       wsb  = ws  + (size_t)b * WSTRIDE;

    __shared__ int histS[HBINS];
    for (int i = tid; i < HBINS; i += blockDim.x) histS[i] = 0;
    __syncthreads();

    v8f acc = {};                         // f32 accumulator, 16x16 tile
    int posC = 0, posmC = 0, negC = 0;

    // per-lane fragment-build constants (hoisted selects)
    const unsigned       chanHi = (Mrow >> 1) & 1;                 // B: pk23 vs pk01
    const unsigned       shiftA = (unsigned)(Mrow & 1) << 4;       // B: hi/lo half
    const int            useB   = (Mrow < 4);
    const unsigned short constH = (Mrow == 4) ? (unsigned short)0x3C00 : (unsigned short)0;

    const int STEPS = 16;                 // 32 px per step per wave
    const int base  = blockIdx.x * (256 * STEPS) + wave * (32 * STEPS);

    for (int s = 0; s < STEPS; ++s) {
        const int p = base + s * 32 + lane;          // coalesced per wave
        const int kid  = gtK[p] & 7;
        const int tidv = gtT[p] & 7;
        const float s0 = simB[p];
        const float s1 = simB[NPIX + p];
        const float s2 = simB[2 * NPIX + p];
        const float s3 = simB[3 * NPIX + p];
        const float sc  = txt[p];
        const float tmv = tmb[p];

        // prefetch next lines of the streamed arrays (global_prefetch_b8)
        __builtin_prefetch(simB + p + 128, 0, 1);
        __builtin_prefetch(txt + p + 128, 0, 1);

        // OHEM statistics on own pixel
        const bool pos = tidv > 0;
        posC  += pos ? 1 : 0;
        posmC += (pos && tmv <= 0.5f) ? 1 : 0;
        negC  += pos ? 0 : 1;
        if (!pos) {
            int bin = (int)fmaf(sc, HSCALE, 512.0f);
            bin = bin < 0 ? 0 : (bin > HBINS - 1 ? HBINS - 1 : bin);
            atomicAdd(&histS[bin], 1);
        }

        // packed one-hot: bits 0-7 kernel id, bits 8-15 text id
        const int mask = (1 << kid) | (256 << tidv);
        H2U u01, u23;
        u01.h = __builtin_amdgcn_cvt_pkrtz(s0, s1);
        u23.h = __builtin_amdgcn_cvt_pkrtz(s2, s3);
        const int pk01 = (int)u01.u;
        const int pk23 = (int)u23.u;

        // Build WMMA fragments (A/B layouts per CDNA5 ISA 7.12.2):
        // lane holds row/col = lane%16; elems 0..7 -> K = half*8+e, 8..15 -> 16+half*8+(e-8)
        Frag16 A, B;
        #pragma unroll
        for (int e = 0; e < 16; ++e) {
            const int K = (e < 8) ? (half * 8 + e) : (8 + half * 8 + e);
            const int      mK  = __shfl(mask, K, 32);
            const unsigned p01 = (unsigned)__shfl(pk01, K, 32);
            const unsigned p23 = (unsigned)__shfl(pk23, K, 32);
            A.u[e] = ((mK >> Mrow) & 1) ? (unsigned short)0x3C00 : (unsigned short)0;
            const unsigned selv = chanHi ? p23 : p01;
            const unsigned short hu = (unsigned short)(selv >> shiftA);
            B.u[e] = useB ? hu : constH;
        }
        // D = A x B + C   (EXEC all ones here: loop is convergent)
        acc = __builtin_amdgcn_wmma_f32_16x16x32_f16(
                  false, A.v, false, B.v, (short)0, acc, false, false);
    }

    // Flush C: element (M = j + 8*half, N = lane%16) = acc[j]
    const int Ncol = Mrow;
    #pragma unroll
    for (int j = 0; j < 8; ++j) {
        const int M = j + 8 * half;
        const float v = acc[j];
        if (Ncol < 4) {
            if (M >= 1 && M <= 7) atomicAdd(&wsb[OFF_KSUM + M * 4 + Ncol], v);
        } else if (Ncol == 4) {
            if (M >= 1 && M <= 7) atomicAdd(&wsb[OFF_KCNT + M], v);         // kcount
            if (M >= 9)           atomicAdd(&wsb[OFF_TCNT + (M - 8)], v);   // tcount
        }
    }

    // Wave-reduce OHEM counters
    #pragma unroll
    for (int o = 16; o > 0; o >>= 1) {
        posC  += __shfl_down(posC,  o, 32);
        posmC += __shfl_down(posmC, o, 32);
        negC  += __shfl_down(negC,  o, 32);
    }
    if (lane == 0) {
        atomicAdd((int*)&wsb[OFF_POS],  posC);
        atomicAdd((int*)&wsb[OFF_POSM], posmC);
        atomicAdd((int*)&wsb[OFF_NEG],  negC);
    }

    __syncthreads();
    int* histG = (int*)&wsb[OFF_HIST];
    for (int i = tid; i < HBINS; i += blockDim.x)
        if (histS[i]) atomicAdd(&histG[i], histS[i]);
}

// Mid: centroids, loss_dis, OHEM threshold from histogram. Tiny, serial.
__global__ void pan_mid(float* ws) {
    const int b = blockIdx.x;
    float* wsb = ws + (size_t)b * WSTRIDE;
    if (threadIdx.x != 0) return;

    float G[8][4];
    bool valid[8];
    int nv = 0;
    for (int k = 1; k < 8; ++k) {
        const float kc = wsb[OFF_KCNT + k];
        const float tc = wsb[OFF_TCNT + k];
        const float inv = 1.0f / fmaxf(kc, 1.0f);
        for (int c = 0; c < 4; ++c) {
            G[k][c] = wsb[OFF_KSUM + k * 4 + c] * inv;
            wsb[OFF_G + k * 4 + c] = G[k][c];
        }
        valid[k] = (kc > 0.0f) && (tc > 0.0f);
        nv += valid[k] ? 1 : 0;
    }
    // loss_dis over valid centroid pairs (<=21 pairs)
    float s = 0.0f;
    for (int k = 1; k < 8; ++k)
        for (int l = k + 1; l < 8; ++l)
            if (valid[k] && valid[l]) {
                float d2 = 0.0f;
                for (int c = 0; c < 4; ++c) {
                    const float d = G[k][c] - G[l][c];
                    d2 += d * d;
                }
                const float gn = d2 > 0.0f ? sqrtf(d2) : 0.0f;
                const float m = fmaxf(3.0f - gn, 0.0f);
                s += log1pf(m * m);
            }
    const float denom = (float)(nv * (nv - 1));
    wsb[OFF_LDIS] = (nv > 1) ? s / fmaxf(denom, 1.0f) : 0.0f;
    ((int*)wsb)[OFF_NV] = nv;

    // OHEM threshold: neg_num-th largest negative score via histogram scan
    const int posT = ((int*)wsb)[OFF_POS];
    const int posM = ((int*)wsb)[OFF_POSM];
    const int negT = ((int*)wsb)[OFF_NEG];
    const int posn = posT - posM;
    int negn = posn * 3;
    if (negn > negT) negn = negT;
    const int fb = (posn == 0) || (negn == 0);
    float thr = -1e10f;
    if (negn > 0) {
        const int* h = (const int*)&wsb[OFF_HIST];
        long long cum = 0;
        int bin = 0;
        for (int i = HBINS - 1; i >= 0; --i) {
            cum += h[i];
            if (cum >= negn) { bin = i; break; }
        }
        thr = ((float)bin - 512.0f) * (1.0f / HSCALE);
    }
    wsb[OFF_THR] = thr;
    ((int*)wsb)[OFF_FLB] = fb;
}

// Pass 2: dice sums + per-pixel aggregation loss using centroids.
__global__ void pan_pass2(const float* __restrict__ outs, const int* __restrict__ labs,
                          const float* __restrict__ tmg, float* __restrict__ ws) {
    const int b = blockIdx.y;
    const int tid = threadIdx.x;
    const int wave = tid >> 5, lane = tid & 31;
    float* wsb = ws + (size_t)b * WSTRIDE;

    __shared__ float Gs[32];
    __shared__ float aggS[8][8];        // [wave][instance] — avoids cross-wave LDS contention
    if (tid < 32) Gs[tid] = wsb[OFF_G + tid];
    if (tid < 64) aggS[tid >> 3][tid & 7] = 0.0f;
    __syncthreads();

    const float thr = wsb[OFF_THR];
    const int   fb  = ((int*)wsb)[OFF_FLB];

    const float* txt  = outs + (size_t)(b * 6 + 0) * NPIX;
    const float* ker  = outs + (size_t)(b * 6 + 1) * NPIX;
    const float* simB = outs + (size_t)(b * 6 + 2) * NPIX;
    const int*   gtT  = labs + (size_t)(b * 2 + 0) * NPIX;
    const int*   gtK  = labs + (size_t)(b * 2 + 1) * NPIX;
    const float* tmb  = tmg + (size_t)b * NPIX;

    float at = 0, bt = 0, ct = 0, ak = 0, bk = 0, ck = 0;
    const int PXT = 16;
    for (int i = 0; i < PXT; ++i) {
        const int p = blockIdx.x * (256 * PXT) + i * 256 + tid;   // coalesced
        const float x   = txt[p];
        const float kx  = ker[p];
        const float tmv = tmb[p];
        const int   ti  = gtT[p];
        const int   ki  = gtK[p];
        const float s0 = simB[p];
        const float s1 = simB[NPIX + p];
        const float s2 = simB[2 * NPIX + p];
        const float s3 = simB[3 * NPIX + p];

        const float pt  = 1.0f / (1.0f + __expf(-x));
        const bool  pos = ti > 0;
        const float sel = fb ? tmv : ((((x >= thr) || pos) && (tmv > 0.5f)) ? 1.0f : 0.0f);
        const float tt  = (pos ? 1.0f : 0.0f) * sel;
        const float pp  = pt * sel;
        at += pp * tt; bt += pp * pp; ct += tt * tt;

        const float selk = ((pt > 0.5f) && (tmv > 0.5f)) ? 1.0f : 0.0f;
        const float pk = (1.0f / (1.0f + __expf(-kx))) * selk;
        const float tk = (ki > 0 ? 1.0f : 0.0f) * selk;
        ak += pk * tk; bk += pk * pk; ck += tk * tk;

        if (ti >= 1 && ti <= 7) {
            const float d0 = s0 - Gs[ti * 4 + 0];
            const float d1 = s1 - Gs[ti * 4 + 1];
            const float d2 = s2 - Gs[ti * 4 + 2];
            const float d3 = s3 - Gs[ti * 4 + 3];
            const float sq = d0 * d0 + d1 * d1 + d2 * d2 + d3 * d3;
            const float d  = sq > 0.0f ? sqrtf(sq) : 0.0f;
            const float m  = fmaxf(d - 0.5f, 0.0f);
            atomicAdd(&aggS[wave][ti], log1pf(m * m));
        }
    }

    #pragma unroll
    for (int o = 16; o > 0; o >>= 1) {
        at += __shfl_down(at, o, 32); bt += __shfl_down(bt, o, 32);
        ct += __shfl_down(ct, o, 32); ak += __shfl_down(ak, o, 32);
        bk += __shfl_down(bk, o, 32); ck += __shfl_down(ck, o, 32);
    }
    if (lane == 0) {
        atomicAdd(&wsb[OFF_DT + 0], at);
        atomicAdd(&wsb[OFF_DT + 1], bt);
        atomicAdd(&wsb[OFF_DT + 2], ct);
        atomicAdd(&wsb[OFF_DK + 0], ak);
        atomicAdd(&wsb[OFF_DK + 1], bk);
        atomicAdd(&wsb[OFF_DK + 2], ck);
    }
    __syncthreads();
    if (tid < 64) {
        const float v = aggS[tid >> 3][tid & 7];
        if (v != 0.0f) atomicAdd(&wsb[OFF_AGG + (tid & 7)], v);
    }
}

__global__ void pan_final(const float* __restrict__ ws, float* __restrict__ out) {
    __shared__ float lt[8], lk[8], la[8], ld[8];
    const int b = threadIdx.x;
    if (b < NBATCH) {
        const float* wsb = ws + (size_t)b * WSTRIDE;
        float a  = wsb[OFF_DT + 0], bb = wsb[OFF_DT + 1], c = wsb[OFF_DT + 2];
        lt[b] = 1.0f - 2.0f * a / (bb + c + 2e-3f);
        a = wsb[OFF_DK + 0]; bb = wsb[OFF_DK + 1]; c = wsb[OFF_DK + 2];
        lk[b] = 1.0f - 2.0f * a / (bb + c + 2e-3f);
        const int nv = ((const int*)wsb)[OFF_NV];
        float s = 0.0f;
        for (int k = 1; k < 8; ++k) {
            const float kc = wsb[OFF_KCNT + k];
            const float tc = wsb[OFF_TCNT + k];
            if (kc > 0.0f && tc > 0.0f) s += wsb[OFF_AGG + k] / fmaxf(tc, 1.0f);
        }
        la[b] = s / fmaxf((float)nv, 1.0f);
        ld[b] = wsb[OFF_LDIS];
    }
    __syncthreads();
    if (threadIdx.x == 0) {
        float t = 0, k = 0, ag = 0, di = 0;
        for (int i = 0; i < NBATCH; ++i) { t += lt[i]; k += lk[i]; ag += la[i]; di += ld[i]; }
        t *= 1.0f / NBATCH; k *= 1.0f / NBATCH; ag *= 1.0f / NBATCH; di *= 1.0f / NBATCH;
        out[0] = t + 0.5f * k + 0.25f * (ag + di);  // loss_all
        out[1] = t;
        out[2] = k;
        out[3] = ag;
        out[4] = di;
    }
}

extern "C" void kernel_launch(void* const* d_in, const int* in_sizes, int n_in,
                              void* d_out, int out_size, void* d_ws, size_t ws_size,
                              hipStream_t stream) {
    const float* outs = (const float*)d_in[0];   // [8,6,640,640] f32
    const int*   labs = (const int*)d_in[1];     // [8,2,640,640] i32
    const float* tmg  = (const float*)d_in[2];   // [8,640,640]   f32
    float* out = (float*)d_out;                  // 5 scalars
    float* ws  = (float*)d_ws;

    (void)in_sizes; (void)n_in; (void)out_size; (void)ws_size;

    pan_zero <<<(NBATCH * WSTRIDE) / 256, 256, 0, stream>>>(ws);
    pan_pass1<<<dim3(NPIX / 4096, NBATCH), 256, 0, stream>>>(outs, labs, tmg, ws);
    pan_mid  <<<NBATCH, 32, 0, stream>>>(ws);
    pan_pass2<<<dim3(NPIX / 4096, NBATCH), 256, 0, stream>>>(outs, labs, tmg, ws);
    pan_final<<<1, 32, 0, stream>>>(ws, out);
}
